// HyperDrugRec_59356448030824
// MI455X (gfx1250) — matrix operand order repaired and checked by reading, not compiled, not run on hardware
//
#include <hip/hip_runtime.h>
#include <hip/hip_bf16.h>

// ---------------------------------------------------------------------------
// HyperDrugRec forward for gfx1250 (MI455X).  All dense contractions use
// v_wmma_f32_16x16x32_f16 (f16 inputs, f32 accumulate); each wave computes a
// 16x32 C tile (2 WMMAs sharing one A fragment).  A / B^T operands are loaded
// as contiguous b128 vectors per lane; K x N B tiles are staged through LDS
// once per block (shared by 4 waves).  Everything else is fused per-row
// wave32 VALU work.  Compile-only target.
// ---------------------------------------------------------------------------

typedef __attribute__((ext_vector_type(16))) _Float16 v16h;
typedef __attribute__((ext_vector_type(8)))  float    v8f;

#define DEV __device__ __forceinline__

static constexpr int D    = 128;
static constexpr int H    = 4;
static constexpr int DH   = 32;
static constexpr int Bsz  = 64;
static constexpr int V    = 32;
static constexpr int S    = 32;
static constexpr int BV   = Bsz * V;        // 2048
static constexpr int NE   = 8000;
static constexpr int KNOD = 8;
static constexpr int NNZ  = NE * KNOD;      // 64000
static constexpr int VMED = 131;

// -------------------------------- reductions -------------------------------

DEV float wave_reduce(float v) {
#pragma unroll
  for (int off = 16; off; off >>= 1) v += __shfl_xor(v, off, 32);
  return v;
}

DEV float block_reduce128(float v, float* lds4) {
  v = wave_reduce(v);
  int w = threadIdx.x >> 5;
  if ((threadIdx.x & 31) == 0) lds4[w] = v;
  __syncthreads();
  float r = lds4[0] + lds4[1] + lds4[2] + lds4[3];
  __syncthreads();
  return r;
}

// ---------------------------- WMMA operand packs ---------------------------

DEV v16h zero16() { v16h z = {}; return z; }

// 16 contiguous floats -> f16 fragment (used for B^T rows)
DEV v16h pack16(const float* __restrict__ p) {
  const float4* q = (const float4*)p;
  float4 c0 = q[0], c1 = q[1], c2 = q[2], c3 = q[3];
  v16h v;
  v[0]  = (_Float16)c0.x; v[1]  = (_Float16)c0.y; v[2]  = (_Float16)c0.z; v[3]  = (_Float16)c0.w;
  v[4]  = (_Float16)c1.x; v[5]  = (_Float16)c1.y; v[6]  = (_Float16)c1.z; v[7]  = (_Float16)c1.w;
  v[8]  = (_Float16)c2.x; v[9]  = (_Float16)c2.y; v[10] = (_Float16)c2.z; v[11] = (_Float16)c2.w;
  v[12] = (_Float16)c3.x; v[13] = (_Float16)c3.y; v[14] = (_Float16)c3.z; v[15] = (_Float16)c3.w;
  return v;
}

// A fragment: lane needs row[k0+half*16 .. +16) (contiguous k, matching B)
// -> two b128 pairs, both 16B-aligned at call sites.
DEV v16h pack_a(const float* __restrict__ row_k0, int half) {
  const float4* p0 = (const float4*)(row_k0 + half * 16);
  const float4* p1 = (const float4*)(row_k0 + half * 16 + 8);
  float4 c0 = p0[0], c1 = p0[1], c2 = p1[0], c3 = p1[1];
  v16h v;
  v[0]  = (_Float16)c0.x; v[1]  = (_Float16)c0.y; v[2]  = (_Float16)c0.z; v[3]  = (_Float16)c0.w;
  v[4]  = (_Float16)c1.x; v[5]  = (_Float16)c1.y; v[6]  = (_Float16)c1.z; v[7]  = (_Float16)c1.w;
  v[8]  = (_Float16)c2.x; v[9]  = (_Float16)c2.y; v[10] = (_Float16)c2.z; v[11] = (_Float16)c2.w;
  v[12] = (_Float16)c3.x; v[13] = (_Float16)c3.y; v[14] = (_Float16)c3.z; v[15] = (_Float16)c3.w;
  return v;
}

// -------------------------------- utility ----------------------------------

__global__ void zero_f32(float* __restrict__ p, long n) {
  long i  = (long)blockIdx.x * blockDim.x + threadIdx.x;
  long st = (long)gridDim.x * blockDim.x;
  for (; i < n; i += st) p[i] = 0.0f;
}

__global__ void copy_row128(const float* __restrict__ src, float* __restrict__ dst) {
  dst[threadIdx.x] = src[threadIdx.x];
}

__global__ void add_n(const float* __restrict__ a, const float* __restrict__ b,
                      const float* __restrict__ c, float* __restrict__ out, long n) {
  long i = (long)blockIdx.x * blockDim.x + threadIdx.x;
  if (i < n) {
    float t = a[i] + b[i];
    if (c) t += c[i];
    out[i] = t;
  }
}

__global__ void concat2(const float* __restrict__ a, const float* __restrict__ b,
                        float* __restrict__ out) {  // (rows,128)x2 -> (rows,256)
  long r = blockIdx.x; int t = threadIdx.x;         // 256 threads
  out[r * 256 + t] = (t < 128) ? a[r * 128 + t] : b[r * 128 + (t - 128)];
}

// ------------------------- WMMA GEMM (f16 in, f32 acc) ---------------------
// C[M,N] = A[M,K] * op(B) + bias.  K % 32 == 0 at every call site.
// Block = 4 waves; wave w owns C tile [tm..tm+16) x [tn..tn+32).
// TRANSB: B stored (N,K) row-major -> per-lane contiguous b128 loads.
// else:   B (K,N) row-major -> 32x32 tile staged via LDS once per block.
template <bool TRANSB>
__global__ void gemm_wmma(const float* __restrict__ A, const float* __restrict__ B,
                          const float* __restrict__ bias, float* __restrict__ C,
                          int M, int N, int K, int lda, int ldb, int ldc) {
  __shared__ float bs[32][33];
  const int wave = threadIdx.x >> 5;
  const int lane = threadIdx.x & 31;
  const int half = lane >> 4;
  const int l16  = lane & 15;
  const int tn   = blockIdx.x * 32;
  const int tm   = (blockIdx.y * 4 + wave) * 16;
  const int m    = tm + l16;
  const int n0   = tn + l16;
  const int n1   = n0 + 16;
  const bool mok = (m < M);

  v8f acc0 = {}, acc1 = {};
  for (int k0 = 0; k0 < K; k0 += 32) {
    v16h av = zero16();
    if (mok) av = pack_a(A + (long)m * lda + k0, half);   // row fully valid: K%32==0
    v16h bv0, bv1;
    if (TRANSB) {
      bv0 = (n0 < N) ? pack16(B + (long)n0 * ldb + k0 + half * 16) : zero16();
      bv1 = (n1 < N) ? pack16(B + (long)n1 * ldb + k0 + half * 16) : zero16();
    } else {
      // cooperative, coalesced stage of B[k0..k0+32) x [tn..tn+32) (zero-padded)
      const int row = threadIdx.x >> 2;          // 0..31 (k)
      const int cg  = (threadIdx.x & 3) * 8;     // 0,8,16,24 (n)
      const float* brow = B + (long)(k0 + row) * ldb + tn + cg;
#pragma unroll
      for (int i = 0; i < 8; ++i)
        bs[row][cg + i] = (tn + cg + i < N) ? brow[i] : 0.f;
      __syncthreads();
#pragma unroll
      for (int r = 0; r < 8; ++r) {
        const int kk = 2 * r + half * 16;
        bv0[2 * r]     = (_Float16)bs[kk][l16];
        bv0[2 * r + 1] = (_Float16)bs[kk + 1][l16];
        bv1[2 * r]     = (_Float16)bs[kk][l16 + 16];
        bv1[2 * r + 1] = (_Float16)bs[kk + 1][l16 + 16];
      }
    }
    acc0 = __builtin_amdgcn_wmma_f32_16x16x32_f16(false, av, false, bv0, (short)0, acc0, false, false);
    acc1 = __builtin_amdgcn_wmma_f32_16x16x32_f16(false, av, false, bv1, (short)0, acc1, false, false);
    if (!TRANSB) __syncthreads();
  }
  const float b0 = (bias && n0 < N) ? bias[n0] : 0.f;
  const float b1 = (bias && n1 < N) ? bias[n1] : 0.f;
  if (tm + 16 <= M && tn + 32 <= N) {            // wave-uniform fast path
#pragma unroll
    for (int r = 0; r < 8; ++r) {
      float* pc = C + (long)(tm + r + half * 8) * ldc + tn;
      pc[l16]      = acc0[r] + b0;
      pc[l16 + 16] = acc1[r] + b1;
    }
  } else {
#pragma unroll
    for (int r = 0; r < 8; ++r) {
      const int mm = tm + r + half * 8;
      if (mm < M) {
        if (n0 < N) C[(long)mm * ldc + n0] = acc0[r] + b0;
        if (n1 < N) C[(long)mm * ldc + n1] = acc1[r] + b1;
      }
    }
  }
}

// Gathered-A WMMA GEMM for node2edge: A(M,32) row r = full[rec[r>>2]] slice
// [(r&3)*32 .. +32), B = lin (32,32) staged once in LDS, C = xh (M,32).
// M % 64 == 0, K == N == 32: fully unguarded.
__global__ void gemm_wmma_gatherA(const float* __restrict__ full, const int* __restrict__ rec,
                                  const float* __restrict__ B, float* __restrict__ C,
                                  int M, int voc) {
  __shared__ float bs[32][33];
  const int wave = threadIdx.x >> 5;
  const int lane = threadIdx.x & 31;
  const int half = lane >> 4;
  const int l16  = lane & 15;
  const int tm   = (blockIdx.x * 4 + wave) * 16;
  {
    const int row = threadIdx.x >> 2;
    const int cg  = (threadIdx.x & 3) * 8;
#pragma unroll
    for (int i = 0; i < 8; ++i) bs[row][cg + i] = B[row * 32 + cg + i];
  }
  const int m = tm + l16;
  int node = rec[m >> 2];
  node = (node < 0) ? 0 : ((node > voc) ? voc : node);
  v16h av = pack_a(full + (long)node * D + (m & 3) * DH, half);
  __syncthreads();
  v16h bv0, bv1;
#pragma unroll
  for (int r = 0; r < 8; ++r) {
    const int kk = 2 * r + half * 16;
    bv0[2 * r]     = (_Float16)bs[kk][l16];
    bv0[2 * r + 1] = (_Float16)bs[kk + 1][l16];
    bv1[2 * r]     = (_Float16)bs[kk][l16 + 16];
    bv1[2 * r + 1] = (_Float16)bs[kk + 1][l16 + 16];
  }
  v8f acc0 = {}, acc1 = {};
  acc0 = __builtin_amdgcn_wmma_f32_16x16x32_f16(false, av, false, bv0, (short)0, acc0, false, false);
  acc1 = __builtin_amdgcn_wmma_f32_16x16x32_f16(false, av, false, bv1, (short)0, acc1, false, false);
#pragma unroll
  for (int r = 0; r < 8; ++r) {
    float* pc = C + (long)(tm + r + half * 8) * 32;
    pc[l16]      = acc0[r];
    pc[l16 + 16] = acc1[r];
  }
}

// ------------------------- hypergraph conv pieces --------------------------

// he[e] = mean over the 8 members (deg_e == 8 structurally)
__global__ void hyperedge_mean(const float* __restrict__ x, const int* __restrict__ nid,
                               float* __restrict__ he) {
  const int e = blockIdx.x, d = threadIdx.x;
  float s = 0.f;
#pragma unroll
  for (int j = 0; j < KNOD; ++j) s += x[(long)nid[e * KNOD + j] * D + d];
  he[(long)e * D + d] = s * 0.125f;
}

__global__ void count_nodes(const int* __restrict__ nid, float* __restrict__ deg, int nnz) {
  int i = blockIdx.x * 256 + threadIdx.x;
  if (i < nnz) atomicAdd(&deg[nid[i]], 1.0f);
}

// per hyperedge: attention logits, leaky-relu(0.2), softmax over 8 members,
// node->edge aggregation (b_inv = 1/8).  256 threads / block.
__global__ void edge_attn(const float* __restrict__ xl, const float* __restrict__ eaW,
                          const float* __restrict__ att, const int* __restrict__ nid,
                          float* __restrict__ alpha, float* __restrict__ out_e) {
  __shared__ int   nd[KNOD];
  __shared__ float araw[KNOD][H];
  __shared__ float aal[KNOD][H];
  const int e = blockIdx.x, tid = threadIdx.x;
  if (tid < KNOD) nd[tid] = nid[e * KNOD + tid];
  __syncthreads();
  const int w = tid >> 5, lane = tid & 31;
#pragma unroll
  for (int q = 0; q < 4; ++q) {               // 8 waves x 4 pairs = 32 (j,h)
    const int p = w * 4 + q, j = p >> 2, h = p & 3;
    float s = 0.f;
    for (int d = lane; d < 256; d += 32) {
      const float v = (d < 128) ? xl[(long)nd[j] * 512 + h * 128 + d]
                                : eaW[(long)e * 512 + h * 128 + (d - 128)];
      s += v * att[h * 256 + d];
    }
    s = wave_reduce(s);
    if (lane == 0) araw[j][h] = (s > 0.f) ? s : 0.2f * s;
  }
  __syncthreads();
  if (tid < H) {                              // softmax over the 8 members
    const int h = tid;
    float mx = -1e30f;
    for (int j = 0; j < KNOD; ++j) mx = fmaxf(mx, araw[j][h]);
    float sum = 0.f;
    for (int j = 0; j < KNOD; ++j) { float ev = __expf(araw[j][h] - mx); aal[j][h] = ev; sum += ev; }
    const float inv = 1.f / sum;
    for (int j = 0; j < KNOD; ++j) aal[j][h] *= inv;
  }
  __syncthreads();
  if (tid < 32) {
    const int j = tid >> 2, h = tid & 3;
    alpha[(long)(e * KNOD + j) * H + h] = aal[j][h];
  }
  for (int idx = tid; idx < 512; idx += 256) {
    const int h = idx >> 7, d = idx & 127;
    float s = 0.f;
#pragma unroll
    for (int j = 0; j < KNOD; ++j) s += xl[(long)nd[j] * 512 + h * 128 + d] * aal[j][h];
    out_e[(long)e * 512 + idx] = s * 0.125f;   // b_inv = 1/8
  }
}

// edge->node scatter with d_inv and head-mean folded in
__global__ void node_scatter(const float* __restrict__ out_e, const float* __restrict__ alpha,
                             const int* __restrict__ nid, const float* __restrict__ deg,
                             float* __restrict__ acc) {
  const int i = blockIdx.x, d = threadIdx.x;
  const int e = i >> 3;
  const int node = nid[i];
  const float dinv = 1.0f / deg[node];         // node occurs in nid => deg >= 1
  float s = 0.f;
#pragma unroll
  for (int h = 0; h < H; ++h) s += out_e[(long)e * 512 + h * 128 + d] * alpha[(long)i * H + h];
  atomicAdd(&acc[(long)node * D + d], s * dinv * 0.25f);  // mean over heads
}

// ------------------------- LayerNorm (row length 128) ----------------------
// out = LN( lrelu( s0 (+s1) (+s2) (+bias) ) ) * g + be ;  slope < 0 => no lrelu
__global__ void fused_ln(const float* __restrict__ s0, const float* __restrict__ s1,
                         const float* __restrict__ s2, const float* __restrict__ bias,
                         const float* __restrict__ g, const float* __restrict__ be,
                         float* __restrict__ out, float slope) {
  __shared__ float red[4];
  const long r = blockIdx.x; const int d = threadIdx.x;
  const long o = r * D + d;
  float t = s0[o];
  if (s1)   t += s1[o];
  if (s2)   t += s2[o];
  if (bias) t += bias[d];
  if (slope >= 0.f) t = (t > 0.f) ? t : slope * t;
  const float m  = block_reduce128(t, red) * (1.f / 128.f);
  const float dv = t - m;
  const float v  = block_reduce128(dv * dv, red) * (1.f / 128.f);
  out[o] = dv * rsqrtf(v + 1e-5f) * g[d] + be[d];
}

// ------------------------- node2edge pooling -------------------------------

__global__ void mean_over_set(const float* __restrict__ xh, float* __restrict__ out) {
  const int bv = blockIdx.x, d = threadIdx.x;
  float s = 0.f;
  for (int j = 0; j < S; ++j) s += xh[((long)bv * S + j) * D + d];
  out[(long)bv * D + d] = s * (1.f / 32.f);
}

__global__ void set_pool(const float* __restrict__ xh, const float* __restrict__ ea2,
                         const float* __restrict__ att2, float* __restrict__ visit) {
  __shared__ float edot[H];
  __shared__ float sc[S][H];
  __shared__ float mh[H], sh_[H];
  const int bv = blockIdx.x, tid = threadIdx.x;   // 128 threads
  if (tid < H) {
    const int h = tid; float s = 0.f;
    for (int dh = 0; dh < DH; ++dh) s += ea2[(long)bv * D + h * DH + dh] * att2[h * 64 + 32 + dh];
    edot[h] = s;
  }
  __syncthreads();
  {
    const int s_ = tid >> 2, h = tid & 3;
    float a = edot[h];
    for (int dh = 0; dh < DH; ++dh) a += xh[((long)bv * S + s_) * D + h * DH + dh] * att2[h * 64 + dh];
    sc[s_][h] = (a > 0.f) ? a : 0.01f * a;
  }
  __syncthreads();
  if (tid < H) {
    const int h = tid; float mx = -1e30f;
    for (int j = 0; j < S; ++j) mx = fmaxf(mx, sc[j][h]);
    float s = 0.f;
    for (int j = 0; j < S; ++j) s += __expf(sc[j][h] - mx);
    mh[h] = mx; sh_[h] = s;
  }
  __syncthreads();
  {
    const int d = tid, h = d >> 5;
    const float inv = 1.f / sh_[h];
    float s = 0.f;
    for (int j = 0; j < S; ++j)
      s += __expf(sc[j][h] - mh[h]) * inv * xh[((long)bv * S + j) * D + d];
    visit[(long)bv * D + d] = s;
  }
}

// ------------------------- MHA (V=32, causal) ------------------------------

__global__ void shift_visits(const float* __restrict__ visit, float* __restrict__ hist) {
  const int bv = blockIdx.x, d = threadIdx.x;
  const int v = bv & 31;
  hist[(long)bv * D + d] = (v == 0) ? 0.f : visit[(long)(bv - 1) * D + d];
}

// one wave per (b,h); thread = query row; mask is structurally causal
__global__ void mha_attn(const float* __restrict__ qkv, float* __restrict__ o) {
  const int bh = blockIdx.x, b = bh >> 2, h = bh & 3;
  const int qi = threadIdx.x;                       // 32 threads
  float qv[DH], p[V];
#pragma unroll
  for (int d = 0; d < DH; ++d) qv[d] = qkv[(long)(b * V + qi) * 384 + h * DH + d];
  float mx = -1e30f;
#pragma unroll
  for (int k = 0; k < V; ++k) {
    float s = 0.f;
#pragma unroll
    for (int d = 0; d < DH; ++d) s += qv[d] * qkv[(long)(b * V + k) * 384 + 128 + h * DH + d];
    s *= 0.17677669529663687f;                      // 1/sqrt(32)
    if (k > qi) s = -1e9f;                          // causal (attn_mask = triu)
    p[k] = s; mx = fmaxf(mx, s);
  }
  float sum = 0.f;
#pragma unroll
  for (int k = 0; k < V; ++k) { p[k] = __expf(p[k] - mx); sum += p[k]; }
  const float inv = 1.f / sum;
#pragma unroll
  for (int d = 0; d < DH; ++d) {
    float s = 0.f;
#pragma unroll
    for (int k = 0; k < V; ++k) s += p[k] * qkv[(long)(b * V + k) * 384 + 256 + h * DH + d];
    o[(long)(b * V + qi) * D + h * DH + d] = s * inv;
  }
}

// ------------------------- DDI bilinear penalty ----------------------------

__global__ void ddi_penalty(const float* __restrict__ logits, const float* __restrict__ ddi,
                            const unsigned char* __restrict__ kpm, float* __restrict__ out_scalar) {
  __shared__ float prob[VMED];
  __shared__ float red[4];
  const int bv = blockIdx.x, tid = threadIdx.x;     // 128 threads
  if (kpm[bv]) return;                              // padded visit: uniform exit
  for (int m = tid; m < VMED; m += 128)
    prob[m] = 1.f / (1.f + __expf(-logits[(long)bv * VMED + m]));
  __syncthreads();
  float s = 0.f;
  for (int m = tid; m < VMED; m += 128) {
    float t = 0.f;
    for (int n = 0; n < VMED; ++n) t += ddi[m * VMED + n] * prob[n];
    s += prob[m] * t;
  }
  s = block_reduce128(s, red);
  if (tid == 0) atomicAdd(out_scalar, 0.0005f * s);
}

// ===========================================================================
//                               host driver
// ===========================================================================

struct EntP {
  const float *emb, *W, *att, *bias, *g1, *b1, *g2, *b2, *ffnW, *ffnb, *lin, *att2;
};
struct MlpP { const float *W1, *b1, *g1, *be1, *W2, *b2, *g2, *be2; };
struct AttnP { const float *Wi, *bi, *Wo, *bo; };

static inline int cdiv(int a, int b) { return (a + b - 1) / b; }

extern "C" void kernel_launch(void* const* d_in, const int* in_sizes, int n_in,
                              void* d_out, int out_size, void* d_ws, size_t ws_size,
                              hipStream_t stream) {
  (void)in_sizes; (void)n_in; (void)out_size; (void)ws_size;
  const int VOCS[3] = {2000, 1400, 131};

  auto F = [&](int i) { return (const float*)d_in[i]; };
  auto I = [&](int i) { return (const int*)d_in[i]; };

  EntP ent[3];
  for (int n = 0; n < 3; ++n) {
    const int b = n * 12;
    ent[n] = { F(b+0), F(b+1), F(b+2), F(b+3), F(b+4), F(b+5),
               F(b+6), F(b+7), F(b+8), F(b+9), F(b+10), F(b+11) };
  }
  MlpP fus = { F(36), F(37), F(38), F(39), F(40), F(41), F(42), F(43) };
  MlpP pat = { F(44), F(45), F(46), F(47), F(48), F(49), F(50), F(51) };
  AttnP dpA  = { F(52), F(53), F(54), F(55) };
  AttnP medA = { F(56), F(57), F(58), F(59) };
  const float* ddi = F(60);
  const int* records[3] = { I(61), I(62), I(63) };
  const int* node_ids[3] = { I(64), I(65), I(66) };
  // d_in[67..69] edge_ids: structurally repeat(arange(NE),8) — exploited directly
  // d_in[70] attn_mask: structurally causal — recomputed in-kernel
  const unsigned char* kpm = (const unsigned char*)d_in[71];

  float* ws = (float*)d_ws;
  float* lg = (float*)d_out;                      // logits (2048 x 131)
  float* neg = lg + (long)BV * VMED;              // batch_neg scalar

  // ---- workspace layout (floats) ----
  const long of_embf[3] = { 0, 2001L * 128, 2001L * 128 + 1401L * 128 };
  const long SZ_EMBF = 2001L * 128 + 1401L * 128 + 132L * 128;     // 452352
  const long of_visit = SZ_EMBF;                                   // 3 x 262144
  const long of_arena = of_visit + 3L * BV * D;                    // 1238784
  float* embf[3]; float* visit[3];
  for (int n = 0; n < 3; ++n) { embf[n] = ws + of_embf[n]; visit[n] = ws + of_visit + (long)n * BV * D; }
  float* A = ws + of_arena;
  // hypergraph-stage arena
  float* he    = A;                    // 8000*128
  float* xl    = A + 1024000;         // 2000*512
  float* eaW   = A + 2048000;         // 8000*512
  float* out_e = A + 6144000;         // 8000*512
  float* alpha = A + 10240000;        // 64000*4
  float* acc   = A + 10496000;        // 2000*128
  float* ln1   = A + 10752000;        // 2000*128
  float* ffn_t = A + 11008000;        // 2000*128
  float* degn  = A + 11264000;        // 2000
  // node2edge-stage arena (reuse)
  float* xh    = A;                    // 262144*32
  float* meanb = A + 8388608;         // 2048*128
  float* ea2   = A + 8650752;         // 2048*128
  // tail-stage arena (reuse)
  float* dpcat  = A;                   // 2048*256
  float* h1     = A + 524288;         // 2048*128
  float* h2     = A + 786432;         // 2048*128
  float* dp     = A + 1048576;
  float* qkv    = A + 1310720;        // 2048*384
  float* obuf   = A + 2097152;
  float* dprep  = A + 2359296;
  float* mhist  = A + 2621440;
  float* mrep   = A + 2883584;
  float* patcat = A + 3145728;        // 2048*256
  float* path   = A + 3670016;
  float* patient= A + 3932160;
  float* psum   = A + 4194304;

  const dim3 b32(32), b128(128), b256(256);

  // ======================= per-entity: hgconv + node2edge ===================
  for (int n = 0; n < 3; ++n) {
    const int voc = VOCS[n];
    const EntP& p = ent[n];
    const int* nid = node_ids[n];

    hyperedge_mean<<<NE, b128, 0, stream>>>(p.emb, nid, he);
    // xl = x @ W  (voc x 512),  eaW = he @ W  (NE x 512)
    gemm_wmma<false><<<dim3(16, cdiv(voc, 64)), b128, 0, stream>>>(p.emb, p.W, nullptr, xl, voc, 512, D, D, 512, 512);
    gemm_wmma<false><<<dim3(16, cdiv(NE, 64)),  b128, 0, stream>>>(he, p.W, nullptr, eaW, NE, 512, D, D, 512, 512);

    zero_f32<<<64, b256, 0, stream>>>(degn, voc);
    zero_f32<<<1024, b256, 0, stream>>>(acc, (long)voc * D);
    count_nodes<<<cdiv(NNZ, 256), b256, 0, stream>>>(nid, degn, NNZ);
    edge_attn<<<NE, b256, 0, stream>>>(xl, eaW, p.att, nid, alpha, out_e);
    node_scatter<<<NNZ, b128, 0, stream>>>(out_e, alpha, nid, degn, acc);

    // x = LN(x + hg + bias); x = LN(x + x@ffnW + ffnb)
    fused_ln<<<voc, b128, 0, stream>>>(p.emb, acc, nullptr, p.bias, p.g1, p.b1, ln1, -1.f);
    gemm_wmma<false><<<dim3(4, cdiv(voc, 64)), b128, 0, stream>>>(ln1, p.ffnW, p.ffnb, ffn_t, voc, D, D, D, D, D);
    fused_ln<<<voc, b128, 0, stream>>>(ln1, ffn_t, nullptr, nullptr, p.g2, p.b2, embf[n], -1.f);
    copy_row128<<<1, b128, 0, stream>>>(p.emb + (long)voc * D, embf[n] + (long)voc * D);  // raw padding row

    // node2edge: xh = full[records] . lin  (gather fused into WMMA A-load)
    gemm_wmma_gatherA<<<(BV * S * H) / 64, b128, 0, stream>>>(embf[n], records[n], p.lin, xh, BV * S * H, voc);
    mean_over_set<<<BV, b128, 0, stream>>>(xh, meanb);
    gemm_wmma<false><<<dim3(1, (BV * H) / 64), b128, 0, stream>>>(meanb, p.lin, nullptr, ea2, BV * H, DH, DH, DH, DH, DH);
    set_pool<<<BV, b128, 0, stream>>>(xh, ea2, p.att2, visit[n]);
  }

  // ============================== tail stages ===============================
  // dp = visit_d + visit_p + mlp2(cat(visit_d, visit_p))
  concat2<<<BV, b256, 0, stream>>>(visit[0], visit[1], dpcat);
  gemm_wmma<false><<<dim3(4, 32), b128, 0, stream>>>(dpcat, fus.W1, fus.b1, h1, BV, D, 2 * D, 2 * D, D, D);
  fused_ln<<<BV, b128, 0, stream>>>(h1, nullptr, nullptr, nullptr, fus.g1, fus.be1, h2, 0.01f);
  gemm_wmma<false><<<dim3(4, 32), b128, 0, stream>>>(h2, fus.W2, fus.b2, h1, BV, D, D, D, D, D);
  fused_ln<<<BV, b128, 0, stream>>>(h1, nullptr, nullptr, nullptr, fus.g2, fus.be2, h2, 0.01f);
  add_n<<<cdiv(BV * D, 256), b256, 0, stream>>>(visit[0], visit[1], h2, dp, (long)BV * D);

  // d_p_rep = MHA(dp)
  gemm_wmma<true><<<dim3(12, 32), b128, 0, stream>>>(dp, dpA.Wi, dpA.bi, qkv, BV, 3 * D, D, D, D, 3 * D);
  mha_attn<<<Bsz * H, b32, 0, stream>>>(qkv, obuf);
  gemm_wmma<true><<<dim3(4, 32), b128, 0, stream>>>(obuf, dpA.Wo, dpA.bo, dprep, BV, D, D, D, D, D);

  // med_rep = MHA(shift(visit_med))
  shift_visits<<<BV, b128, 0, stream>>>(visit[2], mhist);
  gemm_wmma<true><<<dim3(12, 32), b128, 0, stream>>>(mhist, medA.Wi, medA.bi, qkv, BV, 3 * D, D, D, D, 3 * D);
  mha_attn<<<Bsz * H, b32, 0, stream>>>(qkv, obuf);
  gemm_wmma<true><<<dim3(4, 32), b128, 0, stream>>>(obuf, medA.Wo, medA.bo, mrep, BV, D, D, D, D, D);

  // patient = mlp2(cat(d_p_rep, med_rep))
  concat2<<<BV, b256, 0, stream>>>(dprep, mrep, patcat);
  gemm_wmma<false><<<dim3(4, 32), b128, 0, stream>>>(patcat, pat.W1, pat.b1, path, BV, D, 2 * D, 2 * D, D, D);
  fused_ln<<<BV, b128, 0, stream>>>(path, nullptr, nullptr, nullptr, pat.g1, pat.be1, h1, 0.01f);
  gemm_wmma<false><<<dim3(4, 32), b128, 0, stream>>>(h1, pat.W2, pat.b2, path, BV, D, D, D, D, D);
  fused_ln<<<BV, b128, 0, stream>>>(path, nullptr, nullptr, nullptr, pat.g2, pat.be2, patient, 0.01f);

  // logits = (patient + med_rep) @ emb_med[:131].T   -> straight into d_out
  add_n<<<cdiv(BV * D, 256), b256, 0, stream>>>(patient, mrep, nullptr, psum, (long)BV * D);
  gemm_wmma<true><<<dim3(cdiv(VMED, 32), 32), b128, 0, stream>>>(psum, embf[2], nullptr, lg, BV, VMED, D, D, D, VMED);

  // batch_neg = 5e-4 * sum_valid sigmoid(logits)^T DDI sigmoid(logits)
  zero_f32<<<1, b32, 0, stream>>>(neg, 1);
  ddi_penalty<<<BV, b128, 0, stream>>>(lg, ddi, kpm, neg);
}
